// Discriminator_IM_Cat_88064009437971
// MI455X (gfx1250) — matrix operand order repaired and checked by reading, not compiled, not run on hardware
//
#include <hip/hip_runtime.h>
#include <stdint.h>

// ---------------------------------------------------------------------------
// Types for CDNA5 WMMA
// ---------------------------------------------------------------------------
typedef __attribute__((ext_vector_type(16))) __bf16 v16bf;
typedef __attribute__((ext_vector_type(8)))  float  v8f;

#define TOK   16384   // 64*256 tokens
#define STOK  2048    // 8*256 speaker tokens
#define FDIM  128
#define GATES 512
#define STEPS 16384

__device__ __forceinline__ unsigned short f2bf(float f) {
  unsigned u = __float_as_uint(f);
  u += 0x7FFFu + ((u >> 16) & 1u);          // round-to-nearest-even
  return (unsigned short)(u >> 16);
}

// Async global->LDS 16B copy (ASYNCcnt-tracked), with builtin/asm fallback.
__device__ __forceinline__ void async_copy_b128(unsigned lds_off,
                                                const void* gsrc) {
  asm volatile("global_load_async_to_lds_b128 %0, %1, off"
               :: "v"(lds_off), "v"(gsrc) : "memory");
}

__device__ __forceinline__ void wait_async0() {
#if __has_builtin(__builtin_amdgcn_s_wait_asynccnt)
  __builtin_amdgcn_s_wait_asynccnt(0);
#else
  asm volatile("s_wait_asynccnt 0x0" ::: "memory");
#endif
}

// ---------------------------------------------------------------------------
// Prep kernels: fp32 -> zero-padded bf16, bias sums
// ---------------------------------------------------------------------------
__global__ void cast_pad_bf16_kernel(const float* __restrict__ src,
                                     unsigned short* __restrict__ dst,
                                     int rows, int cols, int ldd) {
  int idx = blockIdx.x * blockDim.x + threadIdx.x;
  int total = rows * ldd;
  if (idx >= total) return;
  int r = idx / ldd, c = idx - r * ldd;
  dst[idx] = (c < cols) ? f2bf(src[(size_t)r * cols + c]) : (unsigned short)0;
}

__global__ void bias_sum_kernel(const float* __restrict__ a,
                                const float* __restrict__ b,
                                float* __restrict__ o, int n) {
  int i = blockIdx.x * blockDim.x + threadIdx.x;
  if (i < n) o[i] = a[i] + b[i];
}

// ---------------------------------------------------------------------------
// WMMA GEMM:  C[M,N] = A[M,K](bf16) * W[N,K]^T(bf16) + bias
//   - block = 128 threads (4 waves); each wave computes a 16x64 tile
//   - B panel (64 x K bf16, contiguous since ldb==K) staged once per block
//     into LDS via global_load_async_to_lds_b128; fragments then come from
//     ds_load_b128 (cheap), A fragments double-buffered from global.
//   arep:   0 = identity A rows; 8 = speaker repeat map within 256-row blocks
//   seqmap: 0 = identity out rows; 1 = (B,T)->(T*64+B) permutation
// ---------------------------------------------------------------------------
__global__ __launch_bounds__(128) void wmma_gemm_kernel(
    const unsigned short* __restrict__ A, int lda,
    const unsigned short* __restrict__ B, int ldb,
    const float* __restrict__ bias,
    float* __restrict__ Cf, unsigned short* __restrict__ Cb, int ldc,
    int M, int N, int K, int arep, int seqmap) {
  __shared__ unsigned short sB[64 * 256];   // up to 32 KB B panel

  int wave = threadIdx.x >> 5;
  int lane = threadIdx.x & 31;
  int hlf  = lane >> 4;       // 0: lanes 0-15, 1: lanes 16-31
  int lq   = lane & 15;

  int m0 = blockIdx.x * 64 + wave * 16;
  int n0 = blockIdx.y * 64;

  // ---- async-stage the 64xK B panel (contiguous: ldb == K) ----
  {
    const unsigned short* bpanel = B + (size_t)n0 * ldb;
    int chunks = (64 * K) / 8;              // 16B chunks
    for (int i = threadIdx.x; i < chunks; i += 128) {
      unsigned lds_off = (unsigned)(size_t)&sB[i * 8];
      async_copy_b128(lds_off, bpanel + i * 8);
    }
    wait_async0();
  }
  __syncthreads();

  int abase = m0;
  if (arep) abase = ((m0 >> 8) / arep) * 256 + (m0 & 255);
  const unsigned short* arow = A + (size_t)(abase + lq) * lda;

  v8f acc[4] = {v8f{0,0,0,0,0,0,0,0}, v8f{0,0,0,0,0,0,0,0},
                v8f{0,0,0,0,0,0,0,0}, v8f{0,0,0,0,0,0,0,0}};

  // A-fragment loader (two contiguous 16B chunks per the WMMA A layout)
  auto loadA = [&](int k0) -> v16bf {
    union { uint4 q[2]; v16bf v; } ua;
    ua.q[0] = *(const uint4*)(arow + k0 + hlf * 8);
    ua.q[1] = *(const uint4*)(arow + k0 + 16 + hlf * 8);
    return ua.v;
  };

  v16bf a_cur = loadA(0);
  for (int k0 = 0; k0 < K; k0 += 32) {
    v16bf a_nxt = (k0 + 32 < K) ? loadA(k0 + 32) : a_cur;   // overlap w/ WMMAs
#pragma unroll
    for (int nt = 0; nt < 4; ++nt) {
      const unsigned short* brow =
          &sB[(nt * 16 + lq) * K + k0 + hlf * 8];
      union { uint4 q[2]; v16bf v; } ub;
      ub.q[0] = *(const uint4*)brow;                        // ds_load_b128
      ub.q[1] = *(const uint4*)(brow + 16);
      acc[nt] = __builtin_amdgcn_wmma_f32_16x16x32_bf16(
          false, a_cur, false, ub.v, (short)0, acc[nt], false, false);
    }
    a_cur = a_nxt;
  }

#pragma unroll
  for (int nt = 0; nt < 4; ++nt) {
    int ncol = n0 + nt * 16 + lq;
    float bv = bias ? bias[ncol] : 0.f;
#pragma unroll
    for (int r = 0; r < 8; ++r) {
      int mrow = m0 + r + hlf * 8;
      int orow = seqmap ? ((mrow & 255) * 64 + (mrow >> 8)) : mrow;
      float v = acc[nt][r] + bv;
      if (Cf) Cf[(size_t)orow * ldc + ncol] = v;
      if (Cb) Cb[(size_t)orow * ldc + ncol] = f2bf(v);
    }
  }
}

// ---------------------------------------------------------------------------
// LSTM scan pass for one layer. gates[s,512] precomputed (x@Wih^T + bih+bhh).
// 512 threads; thread j owns gate row j with its Whh row in VGPRs.
// h broadcast through LDS with b128 loads; c lives in registers of j<128.
// ---------------------------------------------------------------------------
__global__ __launch_bounds__(512) void lstm_scan_kernel(
    const float* __restrict__ gates,
    const float* __restrict__ Whh,          // [512,128] fp32
    unsigned short* __restrict__ Hout,      // bf16 [steps,128] or null
    float* __restrict__ Htail, int tail_from,
    int steps) {
  __shared__ __align__(16) float hbuf[FDIM];
  __shared__ float gact[GATES];

  int j = threadIdx.x;
  float w[FDIM];
#pragma unroll
  for (int k = 0; k < FDIM; k += 4) {
    float4 t = *(const float4*)(Whh + (size_t)j * FDIM + k);
    w[k] = t.x; w[k + 1] = t.y; w[k + 2] = t.z; w[k + 3] = t.w;
  }
  float c = 0.f;
  if (j < FDIM) hbuf[j] = 0.f;
  __syncthreads();

  for (int s = 0; s < steps; ++s) {
    // prefetch the (perfectly predictable) gate stream a few steps ahead
    if (s + 8 < steps)
      __builtin_prefetch(&gates[(size_t)(s + 8) * GATES + j], 0, 1);

    float a = gates[(size_t)s * GATES + j];
#pragma unroll
    for (int k = 0; k < FDIM; k += 4) {
      float4 h4 = *(const float4*)&hbuf[k];   // LDS broadcast b128
      a += w[k] * h4.x + w[k + 1] * h4.y + w[k + 2] * h4.z + w[k + 3] * h4.w;
    }
    float act;
    if (j < 256 || j >= 384) {                // i, f, o -> sigmoid
      act = 1.f / (1.f + __expf(-a));
    } else {                                  // g -> tanh
      float e = __expf(2.f * a);
      act = (e - 1.f) / (e + 1.f);
    }
    gact[j] = act;
    __syncthreads();

    if (j < FDIM) {
      float ig = gact[j], fg = gact[FDIM + j];
      float gg = gact[2 * FDIM + j], og = gact[3 * FDIM + j];
      c = fg * c + ig * gg;
      float e = __expf(2.f * c);
      float h = og * ((e - 1.f) / (e + 1.f));
      hbuf[j] = h;
      if (Hout) Hout[(size_t)s * FDIM + j] = f2bf(h);
      if (Htail && s >= tail_from) Htail[(size_t)(s - tail_from) * FDIM + j] = h;
    }
    __syncthreads();
  }
}

// ---------------------------------------------------------------------------
// Head: out[b] = sigmoid(fc2 . relu(fc1 @ h + b1) + b2)
// ---------------------------------------------------------------------------
__global__ __launch_bounds__(128) void head_kernel(
    const float* __restrict__ H, const float* __restrict__ fc1w,
    const float* __restrict__ fc1b, const float* __restrict__ fc2w,
    const float* __restrict__ fc2b, float* __restrict__ out) {
  __shared__ float ybuf[FDIM];
  int b = blockIdx.x, j = threadIdx.x;
  const float* h = H + (size_t)b * FDIM;
  float acc = fc1b[j];
#pragma unroll 4
  for (int k = 0; k < FDIM; ++k) acc += fc1w[(size_t)j * FDIM + k] * h[k];
  ybuf[j] = fmaxf(acc, 0.f) * fc2w[j];
  __syncthreads();
  for (int off = 64; off > 0; off >>= 1) {
    if (j < off) ybuf[j] += ybuf[j + off];
    __syncthreads();
  }
  if (j == 0) {
    float z = ybuf[0] + fc2b[0];
    out[b] = 1.f / (1.f + __expf(-z));
  }
}

// ---------------------------------------------------------------------------
// Launch
// ---------------------------------------------------------------------------
extern "C" void kernel_launch(void* const* d_in, const int* in_sizes, int n_in,
                              void* d_out, int out_size, void* d_ws, size_t ws_size,
                              hipStream_t stream) {
  const float* sp_emo = (const float*)d_in[0];
  const float* li_emo = (const float*)d_in[1];
  const float* sp_dmm = (const float*)d_in[2];
  const float* li_dmm = (const float*)d_in[3];
  // d_in[4] = repeat_interleave (== 8, folded into row maps)
  const float* emo_w  = (const float*)d_in[5];
  const float* emo_b  = (const float*)d_in[6];
  const float* dmm_w  = (const float*)d_in[7];
  const float* dmm_b  = (const float*)d_in[8];
  const float* efus_w = (const float*)d_in[9];
  const float* efus_b = (const float*)d_in[10];
  const float* dfus_w = (const float*)d_in[11];
  const float* dfus_b = (const float*)d_in[12];
  const float* fus_w  = (const float*)d_in[13];
  const float* fus_b  = (const float*)d_in[14];
  const float* Wih    = (const float*)d_in[15];
  const float* Whh    = (const float*)d_in[16];
  const float* bih    = (const float*)d_in[17];
  const float* bhh    = (const float*)d_in[18];
  const float* fc1_w  = (const float*)d_in[19];
  const float* fc1_b  = (const float*)d_in[20];
  const float* fc2_w  = (const float*)d_in[21];
  const float* fc2_b  = (const float*)d_in[22];
  float* out = (float*)d_out;

  char* ws = (char*)d_ws;
  size_t cur = 0;
  auto alloc = [&](size_t bytes) -> char* {
    char* p = ws + cur;
    cur = (cur + bytes + 255) & ~(size_t)255;
    return p;
  };

  unsigned short* liE  = (unsigned short*)alloc((size_t)TOK * 32 * 2);
  unsigned short* spE  = (unsigned short*)alloc((size_t)STOK * 32 * 2);
  unsigned short* liD  = (unsigned short*)alloc((size_t)TOK * 64 * 2);
  unsigned short* spD  = (unsigned short*)alloc((size_t)STOK * 64 * 2);
  unsigned short* wEmo = (unsigned short*)alloc(128 * 32 * 2);
  unsigned short* wDmm = (unsigned short*)alloc(128 * 64 * 2);
  unsigned short* wEf  = (unsigned short*)alloc(128 * 256 * 2);
  unsigned short* wDf  = (unsigned short*)alloc(128 * 256 * 2);
  unsigned short* wFu  = (unsigned short*)alloc(128 * 256 * 2);
  unsigned short* wIh  = (unsigned short*)alloc((size_t)3 * 512 * 128 * 2);
  unsigned short* catE = (unsigned short*)alloc((size_t)TOK * 256 * 2);
  unsigned short* catD = (unsigned short*)alloc((size_t)TOK * 256 * 2);
  unsigned short* catF = (unsigned short*)alloc((size_t)TOK * 256 * 2);
  unsigned short* encS = (unsigned short*)alloc((size_t)TOK * 128 * 2);
  float*          gts  = (float*)alloc((size_t)TOK * 512 * 4);
  unsigned short* H0   = (unsigned short*)alloc((size_t)TOK * 128 * 2);
  unsigned short* H1   = (unsigned short*)alloc((size_t)TOK * 128 * 2);
  float*          bsum = (float*)alloc(3 * 512 * 4);
  float*          h2t  = (float*)alloc(64 * 128 * 4);

  auto cast = [&](const float* s, unsigned short* d, int rows, int cols, int ldd) {
    int total = rows * ldd;
    cast_pad_bf16_kernel<<<(total + 255) / 256, 256, 0, stream>>>(s, d, rows, cols, ldd);
  };
  cast(li_emo, liE, TOK, 25, 32);
  cast(sp_emo, spE, STOK, 25, 32);
  cast(li_dmm, liD, TOK, 58, 64);
  cast(sp_dmm, spD, STOK, 58, 64);
  cast(emo_w, wEmo, 128, 25, 32);
  cast(dmm_w, wDmm, 128, 58, 64);
  cast(efus_w, wEf, 128, 256, 256);
  cast(dfus_w, wDf, 128, 256, 256);
  cast(fus_w, wFu, 128, 256, 256);
  cast(Wih, wIh, 3 * 512, 128, 128);
  bias_sum_kernel<<<(1536 + 255) / 256, 256, 0, stream>>>(bih, bhh, bsum, 1536);

  auto gemm = [&](const unsigned short* A, int lda, const unsigned short* B, int ldb,
                  const float* bias, float* Cf, unsigned short* Cb, int ldc,
                  int M, int N, int K, int arep, int seqmap) {
    dim3 grid(M / 64, N / 64);
    wmma_gemm_kernel<<<grid, 128, 0, stream>>>(A, lda, B, ldb, bias, Cf, Cb, ldc,
                                               M, N, K, arep, seqmap);
  };

  // Encoder (concats are written in-place into column halves)
  gemm(liE, 32,  wEmo, 32,  emo_b,  nullptr, catE,       256, TOK, 128, 32,  0, 0);
  gemm(spE, 32,  wEmo, 32,  emo_b,  nullptr, catE + 128, 256, TOK, 128, 32,  8, 0);
  gemm(catE, 256, wEf, 256, efus_b, nullptr, catF,       256, TOK, 128, 256, 0, 0);
  gemm(liD, 64,  wDmm, 64,  dmm_b,  nullptr, catD,       256, TOK, 128, 64,  0, 0);
  gemm(spD, 64,  wDmm, 64,  dmm_b,  nullptr, catD + 128, 256, TOK, 128, 64,  8, 0);
  gemm(catD, 256, wDf, 256, dfus_b, nullptr, catF + 128, 256, TOK, 128, 256, 0, 0);
  // enc, permuted to sequence order (t*64 + b)
  gemm(catF, 256, wFu, 256, fus_b,  nullptr, encS,       128, TOK, 128, 256, 0, 1);

  // Layer 0: gates = enc @ Wih0^T + (bih0+bhh0), then serial scan
  gemm(encS, 128, wIh, 128, bsum, gts, nullptr, 512, TOK, 512, 128, 0, 0);
  lstm_scan_kernel<<<1, 512, 0, stream>>>(gts, Whh, H0, nullptr, 1 << 30, STEPS);

  // Layer 1
  gemm(H0, 128, wIh + 512 * 128, 128, bsum + 512, gts, nullptr, 512, TOK, 512, 128, 0, 0);
  lstm_scan_kernel<<<1, 512, 0, stream>>>(gts, Whh + 512 * 128, H1, nullptr, 1 << 30, STEPS);

  // Layer 2 (only last 64 hidden states needed)
  gemm(H1, 128, wIh + 2 * 512 * 128, 128, bsum + 1024, gts, nullptr, 512, TOK, 512, 128, 0, 0);
  lstm_scan_kernel<<<1, 512, 0, stream>>>(gts, Whh + 2 * 512 * 128, nullptr, h2t,
                                          STEPS - 64, STEPS);

  // Head
  head_kernel<<<64, 128, 0, stream>>>(h2t, fc1_w, fc1_b, fc2_w, fc2_b, out);
}
